// VisionTransformerClassifier_26852135535028
// MI455X (gfx1250) — compile-verified, matrix operand order checked by reading, not run on hardware
//
#include <hip/hip_runtime.h>
#include <hip/hip_bf16.h>

// ---------------- model constants ----------------
#define Bc   32
#define Cc   3
#define HWc  224
#define NPc  14
#define PSc  16
#define NPPc 196          // NP*NP
#define Sc   197
#define Dc   768
#define Hc   12
#define DHc  64
#define Lc   12
#define FFc  3072
#define NCLSc 1000

typedef unsigned short u16;

// ---------------- WMMA types ----------------
typedef __attribute__((ext_vector_type(16))) __bf16 v16bf;
typedef __attribute__((ext_vector_type(8)))  float  v8f;
typedef __attribute__((ext_vector_type(4)))  int    v4i;

#if __has_builtin(__builtin_amdgcn_global_load_async_to_lds_b128)
#define HAVE_ASYNC_LDS 1
typedef __attribute__((address_space(1))) v4i* gv4i_p;   // global 128-bit chunk
typedef __attribute__((address_space(3))) v4i* lv4i_p;   // LDS 128-bit chunk
#endif

__device__ __forceinline__ u16 f2bf(float f) {
    union { float f; unsigned int u; } c; c.f = f;
    unsigned int u = c.u;
    u += 0x7fffu + ((u >> 16) & 1u);      // round-to-nearest-even
    return (u16)(u >> 16);
}

__device__ __forceinline__ void wait_async0() {
#ifdef HAVE_ASYNC_LDS
#if __has_builtin(__builtin_amdgcn_s_wait_asynccnt)
    __builtin_amdgcn_s_wait_asynccnt(0);
#else
    asm volatile("s_wait_asynccnt 0x0" ::: "memory");
#endif
#endif
}

enum { GF_ACCUM = 1, GF_RELU = 2, GF_TRANSC = 4, GF_OUTBF16 = 8 };

#define TM 64
#define TN 64
#define TK 32

// Stage one 64x32 bf16 tile -> LDS.
// Fast path: 2x async global->LDS b128 copies per thread (ASYNCcnt-tracked) when
// available, else 2x (global_load_b128 + ds_store_b128).
__device__ __forceinline__ void stage_tile(const u16* __restrict__ P, long long ld,
                                           int row0, int nrows, int k0, int K,
                                           bool vec, u16* __restrict__ dst, int tid)
{
    if (vec && (k0 + TK <= K) && (row0 + TM <= nrows)) {
#pragma unroll
        for (int it = 0; it < 2; ++it) {
            int c = tid + it * 128;              // 256 chunks of 8 halfs
            int m = c >> 2, ko = (c & 3) * 8;
            const u16* src = P + (long long)(row0 + m) * ld + k0 + ko;
            u16* d = dst + m * TK + ko;
#ifdef HAVE_ASYNC_LDS
            __builtin_amdgcn_global_load_async_to_lds_b128(
                (gv4i_p)(size_t)(const void*)src,
                (lv4i_p)(size_t)(void*)d,
                0, 0);
#else
            *(uint4*)d = *(const uint4*)src;
#endif
        }
    } else {
#pragma unroll
        for (int it = 0; it < 2; ++it) {
            int c = tid + it * 128;
            int m = c >> 2, ko = (c & 3) * 8;
            int gm = row0 + m;
            uint4 pk; pk.x = 0; pk.y = 0; pk.z = 0; pk.w = 0;
            u16* e = (u16*)&pk;
            if (gm < nrows) {
                const u16* p = P + (long long)gm * ld + k0 + ko;
                int rem = K - (k0 + ko);
                if (rem > 8) rem = 8;
                for (int t = 0; t < 8; ++t) if (t < rem) e[t] = p[t];
            }
            *(uint4*)(dst + m * TK + ko) = pk;
        }
    }
}

// Generalized batched GEMM:  C[b] (+)= alpha * A[b] @ W[b]^T + bias[b]
//   A, W in bf16 ([M,K] lda / [N,K] ldw row-major).  C fp32, or bf16 if GF_OUTBF16.
//   batch index bz -> (bo = bz / bMod, bi = bz % bMod);
//   A off = bo*aO + bi*aI, W off = bo*wO + bi*wI, bias off = bi*bI, C off = bo*cO + bi*cI.
__global__ __launch_bounds__(128)
void gemm_wmma_bf16(const u16* __restrict__ A, const u16* __restrict__ W,
                    const float* __restrict__ bias, void* __restrict__ Cv,
                    int M, int N, int K, int lda, int ldw, int ldc, int bMod,
                    long long aO, long long aI, long long wO, long long wI,
                    long long bI, long long cO, long long cI,
                    float alpha, int flags)
{
    __shared__ __align__(16) u16 As[2][TM * TK];
    __shared__ __align__(16) u16 Ws[2][TN * TK];

    const int bz = blockIdx.z;
    const int bo = bz / bMod, bi2 = bz % bMod;
    const u16* Ab = A + bo * aO + bi2 * aI;
    const u16* Wb = W + bo * wO + bi2 * wI;
    const float* Bb = bias ? (bias + bi2 * bI) : nullptr;
    const long long cOff = bo * cO + bi2 * cI;

    const int tm = blockIdx.y * TM;
    const int tn = blockIdx.x * TN;
    const int tid  = threadIdx.x;
    const int wave = tid >> 5;
    const int lane = tid & 31;

    const bool vecA = ((lda & 7) == 0) &&
                      ((((unsigned long long)(size_t)(const void*)Ab) & 15ull) == 0ull);
    const bool vecW = ((ldw & 7) == 0) &&
                      ((((unsigned long long)(size_t)(const void*)Wb) & 15ull) == 0ull);

    v8f acc[4] = {};

    const int mrow = (wave << 4) + (lane & 15);
    const int kb   = (lane >> 4) * 8;    // lanes 0-15: K base 0, lanes 16-31: K base 8

    const int nk = (K + TK - 1) / TK;
    stage_tile(Ab, lda, tm, M, 0, K, vecA, As[0], tid);
    stage_tile(Wb, ldw, tn, N, 0, K, vecW, Ws[0], tid);

    for (int kt = 0; kt < nk; ++kt) {
        wait_async0();          // previous round of async tile copies landed in LDS
        __syncthreads();
        const int cur = kt & 1;
        if (kt + 1 < nk) {
            stage_tile(Ab, lda, tm, M, (kt + 1) * TK, K, vecA, As[cur ^ 1], tid);
            stage_tile(Wb, ldw, tn, N, (kt + 1) * TK, K, vecW, Ws[cur ^ 1], tid);
        }
        const u16* Ac = As[cur];
        const u16* Wc = Ws[cur];

        union { v16bf v; uint4 q[2]; } fa;
        fa.q[0] = *(const uint4*)&Ac[mrow * TK + kb];
        fa.q[1] = *(const uint4*)&Ac[mrow * TK + kb + 16];
#pragma unroll
        for (int j = 0; j < 4; ++j) {
            int nrow = (j << 4) + (lane & 15);
            union { v16bf v; uint4 q[2]; } fb;
            fb.q[0] = *(const uint4*)&Wc[nrow * TK + kb];
            fb.q[1] = *(const uint4*)&Wc[nrow * TK + kb + 16];
            acc[j] = __builtin_amdgcn_wmma_f32_16x16x32_bf16(
                false, fa.v, false, fb.v, (short)0, acc[j], false, false);
        }
    }

    float* Cf = (float*)Cv + cOff;
    u16*   Ch = (u16*)Cv + cOff;
    const int mbase = (wave << 4) + ((lane >> 4) << 3);
#pragma unroll
    for (int j = 0; j < 4; ++j) {
        int gn = tn + (j << 4) + (lane & 15);
        if (gn >= N) continue;
        float bvv = Bb ? Bb[gn] : 0.0f;
#pragma unroll
        for (int r = 0; r < 8; ++r) {
            int gm = tm + mbase + r;
            if (gm >= M) continue;
            float val = acc[j][r] * alpha + bvv;
            if (flags & GF_RELU) val = fmaxf(val, 0.0f);
            long long ci = (flags & GF_TRANSC) ? ((long long)gn * ldc + gm)
                                               : ((long long)gm * ldc + gn);
            if (flags & GF_OUTBF16)     Ch[ci] = f2bf(val);
            else if (flags & GF_ACCUM)  Cf[ci] += val;
            else                        Cf[ci] = val;
        }
    }
}

// ---------------- fp32 -> bf16 bulk convert (vectorized) ----------------
__global__ __launch_bounds__(256)
void cvt_bf16_kernel(const float4* __restrict__ in, uint2* __restrict__ out, long long nquads)
{
    long long i = (long long)blockIdx.x * 256 + threadIdx.x;
    long long stride = (long long)gridDim.x * 256;
    for (; i < nquads; i += stride) {
        float4 v = in[i];
        uint2 pk;
        pk.x = (unsigned)f2bf(v.x) | ((unsigned)f2bf(v.y) << 16);
        pk.y = (unsigned)f2bf(v.z) | ((unsigned)f2bf(v.w) << 16);
        out[i] = pk;
    }
}

// ---------------- LayerNorm over last dim, bf16 output ----------------
__global__ __launch_bounds__(256)
void ln_kernel(const float* __restrict__ x, const float* __restrict__ g,
               const float* __restrict__ be, u16* __restrict__ out, int Dd)
{
    long long row = blockIdx.x;
    const float* xr = x + row * Dd;
    u16* orow = out + row * Dd;
    __shared__ float red[256];
    int t = threadIdx.x;
    float s = 0.0f;
    for (int i = t; i < Dd; i += 256) s += xr[i];
    red[t] = s; __syncthreads();
    for (int o = 128; o > 0; o >>= 1) { if (t < o) red[t] += red[t + o]; __syncthreads(); }
    float mean = red[0] / (float)Dd;
    __syncthreads();
    float v = 0.0f;
    for (int i = t; i < Dd; i += 256) { float d = xr[i] - mean; v += d * d; }
    red[t] = v; __syncthreads();
    for (int o = 128; o > 0; o >>= 1) { if (t < o) red[t] += red[t + o]; __syncthreads(); }
    float rstd = rsqrtf(red[0] / (float)Dd + 1e-5f);
    for (int i = t; i < Dd; i += 256) orow[i] = f2bf((xr[i] - mean) * rstd * g[i] + be[i]);
}

// ---------------- row softmax fp32 in place (final logits) ----------------
__global__ __launch_bounds__(256)
void softmax_rows(float* __restrict__ x, int Len)
{
    float* r = x + (long long)blockIdx.x * Len;
    __shared__ float red[256];
    int t = threadIdx.x;
    float m = -3.4e38f;
    for (int i = t; i < Len; i += 256) m = fmaxf(m, r[i]);
    red[t] = m; __syncthreads();
    for (int o = 128; o > 0; o >>= 1) { if (t < o) red[t] = fmaxf(red[t], red[t + o]); __syncthreads(); }
    m = red[0]; __syncthreads();
    float s = 0.0f;
    for (int i = t; i < Len; i += 256) { float e = __expf(r[i] - m); r[i] = e; s += e; }
    red[t] = s; __syncthreads();
    for (int o = 128; o > 0; o >>= 1) { if (t < o) red[t] += red[t + o]; __syncthreads(); }
    float inv = 1.0f / red[0];
    for (int i = t; i < Len; i += 256) r[i] *= inv;
}

// ---------------- row softmax: fp32 scores -> bf16 probabilities ----------------
__global__ __launch_bounds__(256)
void softmax_bf16(float* __restrict__ sc, u16* __restrict__ pr, int Len)
{
    float* r = sc + (long long)blockIdx.x * Len;
    u16*   o = pr + (long long)blockIdx.x * Len;
    __shared__ float red[256];
    int t = threadIdx.x;
    float m = -3.4e38f;
    for (int i = t; i < Len; i += 256) m = fmaxf(m, r[i]);
    red[t] = m; __syncthreads();
    for (int oo = 128; oo > 0; oo >>= 1) { if (t < oo) red[t] = fmaxf(red[t], red[t + oo]); __syncthreads(); }
    m = red[0]; __syncthreads();
    float s = 0.0f;
    for (int i = t; i < Len; i += 256) { float e = __expf(r[i] - m); r[i] = e; s += e; }
    red[t] = s; __syncthreads();
    for (int oo = 128; oo > 0; oo >>= 1) { if (t < oo) red[t] += red[t + oo]; __syncthreads(); }
    float inv = 1.0f / red[0];
    for (int i = t; i < Len; i += 256) o[i] = f2bf(r[i] * inv);
}

// ---------------- patchify -> bf16 [B,196,768], (c,py,px) flatten ----------------
__global__ __launch_bounds__(256)
void patchify_kernel(const float* __restrict__ img, u16* __restrict__ p, long long total)
{
    long long i = (long long)blockIdx.x * 256 + threadIdx.x;
    if (i >= total) return;
    int d = (int)(i % Dc);
    long long t = i / Dc;
    int pidx = (int)(t % NPPc);
    long long n = t / NPPc;
    int c  = d >> 8;
    int py = (d >> 4) & 15;
    int px = d & 15;
    int pr = pidx / NPc, pc = pidx % NPc;
    p[i] = f2bf(img[((n * Cc + c) * HWc + (pr * PSc + py)) * HWc + (pc * PSc + px)]);
}

// ---------------- cls token + positional embedding (x fp32) ----------------
__global__ __launch_bounds__(256)
void add_pos_cls_kernel(float* __restrict__ x, const float* __restrict__ cls,
                        const float* __restrict__ pos, long long total)
{
    long long i = (long long)blockIdx.x * 256 + threadIdx.x;
    if (i >= total) return;
    int d = (int)(i % Dc);
    int s = (int)((i / Dc) % Sc);
    if (s == 0) x[i] = cls[d] + pos[d];
    else        x[i] += pos[(long long)s * Dc + d];
}

// ---------------- extract cls rows of x -> bf16 [B, D] ----------------
__global__ __launch_bounds__(256)
void cls_extract_kernel(const float* __restrict__ x, u16* __restrict__ xc)
{
    int i = blockIdx.x * 256 + threadIdx.x;
    if (i >= Bc * Dc) return;
    int n = i / Dc, d = i % Dc;
    xc[i] = f2bf(x[(long long)n * Sc * Dc + d]);
}

// ---------------- host side ----------------
static inline void launch_gemm(hipStream_t st, const u16* A, const u16* W,
                               const float* bias, void* C,
                               int M, int N, int K, int lda, int ldw, int ldc,
                               int batch, int bMod,
                               long long aO, long long aI, long long wO, long long wI,
                               long long bI, long long cO, long long cI,
                               float alpha, int flags)
{
    dim3 grid((unsigned)((N + TN - 1) / TN), (unsigned)((M + TM - 1) / TM), (unsigned)batch);
    gemm_wmma_bf16<<<grid, 128, 0, st>>>(A, W, bias, C, M, N, K, lda, ldw, ldc, bMod,
                                         aO, aI, wO, wI, bI, cO, cI, alpha, flags);
}

static inline void launch_cvt(hipStream_t st, const float* in, u16* out, long long n)
{
    long long nq = n >> 2;                       // all sizes are multiples of 4
    unsigned grid = (unsigned)((nq + 255) / 256);
    if (grid > 32768u) grid = 32768u;
    cvt_bf16_kernel<<<grid, 256, 0, st>>>((const float4*)in, (uint2*)out, nq);
}

extern "C" void kernel_launch(void* const* d_in, const int* in_sizes, int n_in,
                              void* d_out, int out_size, void* d_ws, size_t ws_size,
                              hipStream_t stream)
{
    (void)in_sizes; (void)n_in; (void)out_size; (void)ws_size;

    const float* images = (const float*)d_in[0];
    const float* pos    = (const float*)d_in[1];
    const float* W_map  = (const float*)d_in[2];
    const float* b_map  = (const float*)d_in[3];
    const float* cls    = (const float*)d_in[4];
    const float* ln1_g  = (const float*)d_in[5];
    const float* ln1_b  = (const float*)d_in[6];
    const float* Wq     = (const float*)d_in[7];
    const float* bq     = (const float*)d_in[8];
    const float* Wk     = (const float*)d_in[9];
    const float* bk     = (const float*)d_in[10];
    const float* Wv     = (const float*)d_in[11];
    const float* bv     = (const float*)d_in[12];
    const float* ln2_g  = (const float*)d_in[13];
    const float* ln2_b  = (const float*)d_in[14];
    const float* W1     = (const float*)d_in[15];
    const float* b1     = (const float*)d_in[16];
    const float* W2     = (const float*)d_in[17];
    const float* b2     = (const float*)d_in[18];
    const float* Wh     = (const float*)d_in[19];
    const float* bh     = (const float*)d_in[20];
    float* out = (float*)d_out;

    // ---- workspace layout ----
    const long long XN  = (long long)Bc * Sc * Dc;        // 4,841,472
    const long long SCN = (long long)Bc * Hc * Sc * Sc;   // 14,902,656
    const long long QKVW = (long long)Lc * Hc * DHc * DHc;
    const long long FFW  = (long long)Lc * FFc * Dc;

    char* base = (char*)d_ws;
    size_t off = 0;
    auto alloc = [&](size_t bytes) -> void* {
        void* p = base + off;
        off = (off + bytes + 255) & ~(size_t)255;
        return p;
    };

    float* x    = (float*)alloc(XN * 4);                  // [B,S,D] fp32 residual stream
    u16*   h    = (u16*)  alloc(XN * 2);                  // LN output bf16
    u16*   q    = (u16*)  alloc(XN * 2);                  // [B,H,S,DH]
    u16*   k    = (u16*)  alloc(XN * 2);                  // [B,H,S,DH]
    u16*   vT   = (u16*)  alloc(XN * 2);                  // [B,H,DH,S]
    // shared big region: attention uses scores(fp32)+probs(bf16); MLP uses ff; prologue uses patches
    char*  region = (char*)alloc(SCN * 4 + SCN * 2);
    float* scores = (float*)region;                       // [B,H,S,S]
    u16*   probs  = (u16*)(region + SCN * 4);             // [B,H,S,S]
    u16*   ff     = (u16*)region;                         // [B,S,FF] (<= region size)
    u16*   patches= (u16*)region;                         // [B,196,768]
    u16*   xcls   = (u16*)alloc((long long)Bc * Dc * 2);  // [B,D]
    // bf16 weights
    u16* wbmap = (u16*)alloc((long long)Dc * Dc * 2);
    u16* wbq   = (u16*)alloc(QKVW * 2);
    u16* wbk   = (u16*)alloc(QKVW * 2);
    u16* wbv   = (u16*)alloc(QKVW * 2);
    u16* wb1   = (u16*)alloc(FFW * 2);
    u16* wb2   = (u16*)alloc(FFW * 2);
    u16* wbh   = (u16*)alloc((long long)NCLSc * Dc * 2);

    const float scale = 0.125f;   // 1/sqrt(64)

    // 0) one-time weight conversion fp32 -> bf16
    launch_cvt(stream, W_map, wbmap, (long long)Dc * Dc);
    launch_cvt(stream, Wq,    wbq,   QKVW);
    launch_cvt(stream, Wk,    wbk,   QKVW);
    launch_cvt(stream, Wv,    wbv,   QKVW);
    launch_cvt(stream, W1,    wb1,   FFW);
    launch_cvt(stream, W2,    wb2,   FFW);
    launch_cvt(stream, Wh,    wbh,   (long long)NCLSc * Dc);

    // 1) patchify (bf16)
    {
        long long total = (long long)Bc * NPPc * Dc;
        patchify_kernel<<<(unsigned)((total + 255) / 256), 256, 0, stream>>>(images, patches, total);
    }
    // 2) patch embedding -> x rows 1..196 per image (fp32 out, bias included)
    launch_gemm(stream, patches, wbmap, b_map, x + Dc,
                NPPc, Dc, Dc, Dc, Dc, Dc,
                Bc, 1,
                (long long)NPPc * Dc, 0, 0, 0, 0,
                (long long)Sc * Dc, 0, 1.0f, 0);
    // 3) cls token + positional embedding
    add_pos_cls_kernel<<<(unsigned)((XN + 255) / 256), 256, 0, stream>>>(x, cls, pos, XN);

    for (int l = 0; l < Lc; ++l) {
        // LN1 -> h (bf16)
        ln_kernel<<<(unsigned)(Bc * Sc), 256, 0, stream>>>(x, ln1_g + l * Dc, ln1_b + l * Dc, h, Dc);

        // Q, K projections (bf16 out)  [per (n,h): (S,DH) = (S,DH) @ (DH,DH)^T + b]
        launch_gemm(stream, h, wbq + (long long)l * Hc * DHc * DHc, bq + (long long)l * Hc * DHc, q,
                    Sc, DHc, DHc, Dc, DHc, DHc,
                    Bc * Hc, Hc,
                    (long long)Sc * Dc, DHc, 0, (long long)DHc * DHc, DHc,
                    (long long)Hc * Sc * DHc, (long long)Sc * DHc, 1.0f, GF_OUTBF16);
        launch_gemm(stream, h, wbk + (long long)l * Hc * DHc * DHc, bk + (long long)l * Hc * DHc, k,
                    Sc, DHc, DHc, Dc, DHc, DHc,
                    Bc * Hc, Hc,
                    (long long)Sc * Dc, DHc, 0, (long long)DHc * DHc, DHc,
                    (long long)Hc * Sc * DHc, (long long)Sc * DHc, 1.0f, GF_OUTBF16);
        // V projection, transposed bf16 out: vT[b,h] is [DH,S]
        launch_gemm(stream, h, wbv + (long long)l * Hc * DHc * DHc, bv + (long long)l * Hc * DHc, vT,
                    Sc, DHc, DHc, Dc, DHc, Sc,
                    Bc * Hc, Hc,
                    (long long)Sc * Dc, DHc, 0, (long long)DHc * DHc, DHc,
                    (long long)Hc * DHc * Sc, (long long)DHc * Sc, 1.0f, GF_OUTBF16 | GF_TRANSC);

        // scores = scale * q @ k^T  (fp32 out)
        launch_gemm(stream, q, k, nullptr, scores,
                    Sc, Sc, DHc, DHc, DHc, Sc,
                    Bc * Hc, 1,
                    (long long)Sc * DHc, 0, (long long)Sc * DHc, 0, 0,
                    (long long)Sc * Sc, 0, scale, 0);

        // softmax -> bf16 probabilities
        softmax_bf16<<<(unsigned)(Bc * Hc * Sc), 256, 0, stream>>>(scores, probs, Sc);

        // x += probs @ v   (A @ vT^T), fp32 accumulate into head slice of x
        launch_gemm(stream, probs, vT, nullptr, x,
                    Sc, DHc, Sc, Sc, Sc, Dc,
                    Bc * Hc, Hc,
                    (long long)Hc * Sc * Sc, (long long)Sc * Sc,
                    (long long)Hc * DHc * Sc, (long long)DHc * Sc, 0,
                    (long long)Sc * Dc, DHc, 1.0f, GF_ACCUM);

        // LN2 -> h (bf16)
        ln_kernel<<<(unsigned)(Bc * Sc), 256, 0, stream>>>(x, ln2_g + l * Dc, ln2_b + l * Dc, h, Dc);

        // FF1: relu(h @ W1^T + b1) -> ff (bf16)
        launch_gemm(stream, h, wb1 + (long long)l * FFc * Dc, b1 + (long long)l * FFc, ff,
                    Bc * Sc, FFc, Dc, Dc, Dc, FFc,
                    1, 1, 0, 0, 0, 0, 0, 0, 0, 1.0f, GF_RELU | GF_OUTBF16);

        // FF2: x += ff @ W2^T + b2 (fp32 accumulate)
        launch_gemm(stream, ff, wb2 + (long long)l * Dc * FFc, b2 + (long long)l * Dc, x,
                    Bc * Sc, Dc, FFc, FFc, FFc, Dc,
                    1, 1, 0, 0, 0, 0, 0, 0, 0, 1.0f, GF_ACCUM);
    }

    // head: logits = x[:,0,:] @ Wh^T + bh
    cls_extract_kernel<<<(unsigned)((Bc * Dc + 255) / 256), 256, 0, stream>>>(x, xcls);
    launch_gemm(stream, xcls, wbh, bh, out,
                Bc, NCLSc, Dc, Dc, Dc, NCLSc,
                1, 1, 0, 0, 0, 0, 0, 0, 0, 1.0f, 0);
    // final softmax
    softmax_rows<<<(unsigned)Bc, 256, 0, stream>>>(out, NCLSc);
}
